// DMPNN_4621384810929
// MI455X (gfx1250) — compile-verified, matrix operand order checked by reading, not run on hardware
//
#include <hip/hip_runtime.h>

typedef __bf16 bf16;
typedef __attribute__((ext_vector_type(16))) __bf16 v16bf;
typedef __attribute__((ext_vector_type(8)))  __bf16 v8bf;
typedef __attribute__((ext_vector_type(8)))  float  v8f;

#define N_NODES 50000
#define N_EDGES 1600000
#define D_NODE  128
#define D_EDGE  32
#define UNITS   128
#define STEPS   4

// ---------------------------------------------------------------------------
// helpers
// ---------------------------------------------------------------------------
__device__ __forceinline__ v16bf frag_combine(v8bf lo, v8bf hi) {
  return __builtin_shufflevector(lo, hi, 0, 1, 2, 3, 4, 5, 6, 7,
                                 8, 9, 10, 11, 12, 13, 14, 15);
}

// async copy 16B global -> LDS (GLOBAL_LOAD_ASYNC_TO_LDS_B128, ASYNCcnt).
// saddr must be uniform; goff is a per-lane unsigned byte offset (< 2^31).
__device__ __forceinline__ void async_copy16(uint64_t gbase, uint32_t goff,
                                             void* lptr) {
  uint32_t lds_off = (uint32_t)(uintptr_t)lptr;  // low 32 bits = LDS offset
  asm volatile("global_load_async_to_lds_b128 %0, %1, %2"
               :
               : "v"(lds_off), "v"(goff), "s"(gbase)
               : "memory");
}
__device__ __forceinline__ void async_wait0() {
  asm volatile("s_wait_asynccnt 0" ::: "memory");
}

// convert 8 f32 -> v8bf
__device__ __forceinline__ v8bf cvt8(float4 a, float4 b) {
  v8bf m;
  m[0] = (bf16)a.x; m[1] = (bf16)a.y; m[2] = (bf16)a.z; m[3] = (bf16)a.w;
  m[4] = (bf16)b.x; m[5] = (bf16)b.y; m[6] = (bf16)b.z; m[7] = (bf16)b.w;
  return m;
}
// (a||b) - h  -> v8bf
__device__ __forceinline__ v8bf sub_cvt8(float4 a, float4 b, v8bf h) {
  v8bf m;
  m[0] = (bf16)(a.x - (float)h[0]); m[1] = (bf16)(a.y - (float)h[1]);
  m[2] = (bf16)(a.z - (float)h[2]); m[3] = (bf16)(a.w - (float)h[3]);
  m[4] = (bf16)(b.x - (float)h[4]); m[5] = (bf16)(b.y - (float)h[5]);
  m[6] = (bf16)(b.z - (float)h[6]); m[7] = (bf16)(b.w - (float)h[7]);
  return m;
}

// GEMM core: one wave computes a 16-row x 128-col tile.
// a_tile : LDS, row-major [128][astride] bf16 (rows = M)
// wlds   : LDS, pre-swizzled B fragments [kc][t][lane][16] bf16
template <int NKC>
__device__ __forceinline__ void gemm_tile(const bf16* a_tile, int astride,
                                          const bf16* wlds, v8f acc[8]) {
  const int lane = threadIdx.x & 31;
  const int wave = threadIdx.x >> 5;
  const int lm = lane & 15;
  const int lhi = lane >> 4;
#pragma unroll
  for (int kc = 0; kc < NKC; ++kc) {
    const bf16* ar = a_tile + (wave * 16 + lm) * astride + kc * 32 + lhi * 8;
    v16bf af = frag_combine(*(const v8bf*)ar, *(const v8bf*)(ar + 16));
#pragma unroll
    for (int t = 0; t < 8; ++t) {
      const bf16* br = wlds + ((kc * 8 + t) * 32 + lane) * 16;
      v16bf bfv = frag_combine(*(const v8bf*)br, *(const v8bf*)(br + 8));
      acc[t] = __builtin_amdgcn_wmma_f32_16x16x32_bf16(
          false, af, false, bfv, (short)0, acc[t], false, false);
    }
  }
}

// ---------------------------------------------------------------------------
// weight swizzle: W[K][128] f32 -> per-lane bf16 B-fragments
// out[((kc*8 + t)*32 + lane)*16 + j] = W[kc*32 + (j>>3)*16 + (lane>>4)*8 + (j&7)]
//                                       [t*16 + (lane&15)]
// ---------------------------------------------------------------------------
__global__ void dmpnn_wfrag_kernel(const float* __restrict__ W,
                                   bf16* __restrict__ out, int nkc) {
  int idx = blockIdx.x * blockDim.x + threadIdx.x;
  int total = nkc * 8 * 32 * 16;
  if (idx >= total) return;
  int j = idx & 15;
  int lane = (idx >> 4) & 31;
  int t = (idx >> 9) & 7;
  int kc = idx >> 12;
  int n = t * 16 + (lane & 15);
  int hi = lane >> 4;
  int k = kc * 32 + ((j >> 3) << 4) + hi * 8 + (j & 7);
  out[idx] = (bf16)W[k * UNITS + n];
}

__global__ void dmpnn_zero_kernel(float* __restrict__ p, int n) {
  int i = blockIdx.x * blockDim.x + threadIdx.x;
  int stride = gridDim.x * blockDim.x;
  float4 z = {0.f, 0.f, 0.f, 0.f};
  for (int j = i; j < n / 4; j += stride) ((float4*)p)[j] = z;
}

// ---------------------------------------------------------------------------
// init: h = relu([x_src || e_feat] @ Wi + bi), scatter-add h into agg
// ---------------------------------------------------------------------------
__global__ __launch_bounds__(256) void dmpnn_init_kernel(
    const float* __restrict__ node_feature, const float* __restrict__ edge_feature,
    const int* __restrict__ edge_src, const int* __restrict__ edge_dst,
    const bf16* __restrict__ wi_frag, const float* __restrict__ bi,
    bf16* __restrict__ h, float* __restrict__ agg) {
  __shared__ __align__(16) bf16 a_tile[128 * 168];  // K=160, pad to 168
  __shared__ __align__(16) bf16 wlds[5 * 4096];
  const int e0 = blockIdx.x * 128;

  // async stage Wi fragments (40KB = 2560 x 16B)
  const uint64_t wbase = (uint64_t)(uintptr_t)wi_frag;
  for (int i = threadIdx.x; i < 2560; i += 256)
    async_copy16(wbase, (uint32_t)i * 16u, &wlds[i * 8]);

  // build A tile: [x_src || e_feat] in bf16, vectorized 8 cols/chunk
  {
    int row = threadIdx.x >> 1, half = threadIdx.x & 1;
    int e = e0 + row;
    int s = edge_src[e];
    const float* nf = node_feature + (size_t)s * D_NODE;
    const float* ef = edge_feature + (size_t)e * D_EDGE;
    bf16* ar = a_tile + row * 168;
#pragma unroll
    for (int c8 = half * 10; c8 < half * 10 + 10; ++c8) {  // 20 chunks, K=160
      const float* sp = (c8 < 16) ? (nf + c8 * 8) : (ef + c8 * 8 - D_NODE);
      float4 a0 = *(const float4*)sp;
      float4 a1 = *(const float4*)(sp + 4);
      *(v8bf*)(ar + c8 * 8) = cvt8(a0, a1);
    }
  }
  async_wait0();
  __syncthreads();

  v8f acc[8] = {};
  gemm_tile<5>(a_tile, 168, wlds, acc);

  const int lane = threadIdx.x & 31, wave = threadIdx.x >> 5;
  const int lm = lane & 15, lhi = lane >> 4;
  float bias_v[8];
#pragma unroll
  for (int t = 0; t < 8; ++t) bias_v[t] = bi[t * 16 + lm];
#pragma unroll
  for (int r = 0; r < 8; ++r) {
    int e = e0 + wave * 16 + r + 8 * lhi;
    int d = edge_dst[e];
    bf16* hp = h + (size_t)e * UNITS;
    float* ap = agg + (size_t)d * UNITS;
#pragma unroll
    for (int t = 0; t < 8; ++t) {
      float v = fmaxf(acc[t][r] + bias_v[t], 0.f);
      hp[t * 16 + lm] = (bf16)v;
      atomicAdd(ap + t * 16 + lm, v);
    }
  }
}

// ---------------------------------------------------------------------------
// update: msg = agg_in[src] - h[e^1];  h = relu(msg @ Wu + bu + h)  (in place),
// scatter-add new h into agg_out
// ---------------------------------------------------------------------------
__global__ __launch_bounds__(256) void dmpnn_update_kernel(
    const int* __restrict__ edge_src, const int* __restrict__ edge_dst,
    const bf16* __restrict__ wu_frag, const float* __restrict__ bu,
    bf16* __restrict__ h, const float* __restrict__ agg_in,
    float* __restrict__ agg_out) {
  __shared__ __align__(16) bf16 h_tile[128 * 136];  // K=128, pad to 136
  __shared__ __align__(16) bf16 msg[128 * 136];
  __shared__ __align__(16) bf16 wlds[4 * 4096];
  const int e0 = blockIdx.x * 128;

  // async stage h tile (32KB, padded rows) and Wu fragments (32KB)
  const uint64_t hbase = (uint64_t)(uintptr_t)h;
  for (int i = threadIdx.x; i < 128 * 16; i += 256) {  // 16x16B chunks per row
    int row = i >> 4, ch = i & 15;
    uint32_t goff = (uint32_t)(((size_t)(e0 + row) * UNITS + ch * 8) * sizeof(bf16));
    async_copy16(hbase, goff, &h_tile[row * 136 + ch * 8]);
  }
  const uint64_t wbase = (uint64_t)(uintptr_t)wu_frag;
  for (int i = threadIdx.x; i < 2048; i += 256)
    async_copy16(wbase, (uint32_t)i * 16u, &wlds[i * 8]);
  async_wait0();
  __syncthreads();

  // msg = agg_in[src] - h[rev]; rev edge (e^1) is in-tile (128-aligned tiles)
  {
    int row = threadIdx.x >> 1, half = threadIdx.x & 1;
    int e = e0 + row;
    int s = edge_src[e];
    const float4* ag4 = (const float4*)(agg_in + (size_t)s * UNITS);
    const bf16* hr = &h_tile[(row ^ 1) * 136];
    bf16* mr = &msg[row * 136];
#pragma unroll
    for (int c8 = half * 8; c8 < half * 8 + 8; ++c8) {  // 16 chunks of 8 cols
      float4 a0 = ag4[c8 * 2];
      float4 a1 = ag4[c8 * 2 + 1];
      v8bf hv = *(const v8bf*)(hr + c8 * 8);
      *(v8bf*)(mr + c8 * 8) = sub_cvt8(a0, a1, hv);
    }
  }
  __syncthreads();

  const int lane = threadIdx.x & 31, wave = threadIdx.x >> 5;
  const int lm = lane & 15, lhi = lane >> 4;
  v8f acc[8];
#pragma unroll
  for (int t = 0; t < 8; ++t)
#pragma unroll
    for (int r = 0; r < 8; ++r)
      acc[t][r] = (float)h_tile[(wave * 16 + r + 8 * lhi) * 136 + t * 16 + lm];

  gemm_tile<4>(msg, 136, wlds, acc);

  float bias_v[8];
#pragma unroll
  for (int t = 0; t < 8; ++t) bias_v[t] = bu[t * 16 + lm];
#pragma unroll
  for (int r = 0; r < 8; ++r) {
    int e = e0 + wave * 16 + r + 8 * lhi;
    int d = edge_dst[e];
    bf16* hp = h + (size_t)e * UNITS;
    float* ap = agg_out + (size_t)d * UNITS;
#pragma unroll
    for (int t = 0; t < 8; ++t) {
      float v = fmaxf(acc[t][r] + bias_v[t], 0.f);
      hp[t * 16 + lm] = (bf16)v;
      atomicAdd(ap + t * 16 + lm, v);
    }
  }
}

// ---------------------------------------------------------------------------
// final: out = relu([node_feature || node_message] @ Wf + bf)
// ---------------------------------------------------------------------------
__global__ __launch_bounds__(256) void dmpnn_final_kernel(
    const float* __restrict__ node_feature, const float* __restrict__ agg,
    const bf16* __restrict__ wf_frag, const float* __restrict__ bias_f,
    float* __restrict__ out) {
  __shared__ __align__(16) bf16 a_tile[128 * 264];  // K=256, pad to 264
  __shared__ __align__(16) bf16 wlds[8 * 4096];
  const int n0 = blockIdx.x * 128;

  // async stage Wf fragments (64KB = 4096 x 16B)
  const uint64_t wbase = (uint64_t)(uintptr_t)wf_frag;
  for (int i = threadIdx.x; i < 4096; i += 256)
    async_copy16(wbase, (uint32_t)i * 16u, &wlds[i * 8]);

  {
    int row = threadIdx.x >> 1, half = threadIdx.x & 1;
    int node = n0 + row;
    bf16* ar = a_tile + row * 264;
    if (node < N_NODES) {
      const float* nf = node_feature + (size_t)node * D_NODE;
      const float* ag = agg + (size_t)node * UNITS;
#pragma unroll
      for (int c8 = half * 16; c8 < half * 16 + 16; ++c8) {  // 32 chunks, K=256
        const float* sp = (c8 < 16) ? (nf + c8 * 8) : (ag + (c8 - 16) * 8);
        float4 a0 = *(const float4*)sp;
        float4 a1 = *(const float4*)(sp + 4);
        *(v8bf*)(ar + c8 * 8) = cvt8(a0, a1);
      }
    } else {
      float4 z = {0.f, 0.f, 0.f, 0.f};
      v8bf zb = cvt8(z, z);
#pragma unroll
      for (int c8 = half * 16; c8 < half * 16 + 16; ++c8)
        *(v8bf*)(ar + c8 * 8) = zb;
    }
  }
  async_wait0();
  __syncthreads();

  v8f acc[8] = {};
  gemm_tile<8>(a_tile, 264, wlds, acc);

  const int lane = threadIdx.x & 31, wave = threadIdx.x >> 5;
  const int lm = lane & 15, lhi = lane >> 4;
  float bias_v[8];
#pragma unroll
  for (int t = 0; t < 8; ++t) bias_v[t] = bias_f[t * 16 + lm];
#pragma unroll
  for (int r = 0; r < 8; ++r) {
    int node = n0 + wave * 16 + r + 8 * lhi;
    if (node < N_NODES) {
      float* op = out + (size_t)node * UNITS;
#pragma unroll
      for (int t = 0; t < 8; ++t)
        op[t * 16 + lm] = fmaxf(acc[t][r] + bias_v[t], 0.f);
    }
  }
}

// ---------------------------------------------------------------------------
// launch
// ---------------------------------------------------------------------------
extern "C" void kernel_launch(void* const* d_in, const int* in_sizes, int n_in,
                              void* d_out, int out_size, void* d_ws, size_t ws_size,
                              hipStream_t stream) {
  (void)in_sizes; (void)n_in; (void)out_size; (void)ws_size;
  const float* node_feature = (const float*)d_in[0];
  const float* edge_feature = (const float*)d_in[1];
  const int*   edge_src     = (const int*)d_in[2];
  const int*   edge_dst     = (const int*)d_in[3];
  const float* Wi = (const float*)d_in[4];
  const float* bi = (const float*)d_in[5];
  const float* Wu = (const float*)d_in[6];
  const float* bu = (const float*)d_in[7];
  const float* Wf = (const float*)d_in[8];
  const float* bf_bias = (const float*)d_in[9];
  float* out = (float*)d_out;

  // workspace carve-up (~461 MB): h(bf16) + 2x agg(f32) + weight fragments
  char* ws = (char*)d_ws;
  size_t off = 0;
  auto carve = [&](size_t bytes) -> void* {
    void* p = ws + off;
    off = (off + bytes + 255) & ~(size_t)255;
    return p;
  };
  bf16*  h       = (bf16*)carve((size_t)N_EDGES * UNITS * sizeof(bf16));
  float* aggA    = (float*)carve((size_t)N_NODES * UNITS * sizeof(float));
  float* aggB    = (float*)carve((size_t)N_NODES * UNITS * sizeof(float));
  bf16*  wi_frag = (bf16*)carve((size_t)5 * 4096 * sizeof(bf16));
  bf16*  wu_frag = (bf16*)carve((size_t)4 * 4096 * sizeof(bf16));
  bf16*  wf_frag = (bf16*)carve((size_t)8 * 4096 * sizeof(bf16));

  // weight swizzle (once per launch; cheap)
  dmpnn_wfrag_kernel<<<(5 * 4096 + 255) / 256, 256, 0, stream>>>(Wi, wi_frag, 5);
  dmpnn_wfrag_kernel<<<(4 * 4096 + 255) / 256, 256, 0, stream>>>(Wu, wu_frag, 4);
  dmpnn_wfrag_kernel<<<(8 * 4096 + 255) / 256, 256, 0, stream>>>(Wf, wf_frag, 8);

  const int agg_elems = N_NODES * UNITS;
  float* agg_cur = aggA;
  float* agg_nxt = aggB;

  dmpnn_zero_kernel<<<2048, 256, 0, stream>>>(agg_cur, agg_elems);
  dmpnn_init_kernel<<<N_EDGES / 128, 256, 0, stream>>>(
      node_feature, edge_feature, edge_src, edge_dst, wi_frag, bi, h, agg_cur);

  for (int s = 0; s < STEPS; ++s) {
    dmpnn_zero_kernel<<<2048, 256, 0, stream>>>(agg_nxt, agg_elems);
    dmpnn_update_kernel<<<N_EDGES / 128, 256, 0, stream>>>(
        edge_src, edge_dst, wu_frag, bu, h, agg_cur, agg_nxt);
    float* tmp = agg_cur; agg_cur = agg_nxt; agg_nxt = tmp;
  }

  dmpnn_final_kernel<<<(N_NODES + 127) / 128, 256, 0, stream>>>(
      node_feature, agg_cur, wf_frag, bf_bias, out);
}
